// GraphConv_14001593385270
// MI455X (gfx1250) — matrix operand order repaired and checked by reading, not compiled, not run on hardware
//
#include <hip/hip_runtime.h>

typedef __attribute__((ext_vector_type(16))) _Float16 v16h;
typedef __attribute__((ext_vector_type(8)))  _Float16 v8h;
typedef __attribute__((ext_vector_type(4)))  _Float16 v4h;
typedef __attribute__((ext_vector_type(8)))  float    v8f;
typedef __attribute__((ext_vector_type(4)))  float    v4f;

// Problem constants (match reference)
#define GB 4
#define GN 100000
#define GF 64
#define GK 5
#define GOUT 64
#define GNODES (GB * GN)        // 400000
#define FEAT (6 * GF)           // 384
#define KSTEPS (FEAT / 32)      // 12
#define NTILES (GOUT / 16)      // 4
#define MTILES 2                // 2 x 16-row tiles per wave -> M=32 per wave
#define TILE_M 32               // nodes per wave
#define WAVES_PER_BLOCK 2
#define ROW_PAD 8               // row stride 392 halves = 784B (bank-conflict-free frag loads)
#define ROW_STRIDE (FEAT + ROW_PAD)

// ---------------------------------------------------------------------------
// Prep: convert W [64, 384] fp32 -> f16 B-fragments in WMMA VGPR layout.
// Fragment f = s*4 + nt covers K = 32*s..32*s+31, N = nt*16..nt*16+15.
// ISA 16-bit B 32x16 layout: lane l -> column n = l&15;
// lanes 0-15 hold K = 0..7,16..23 of the step, lanes 16-31 hold K = 8..15,24..31.
// Stored as wfrag[f*512 + lane*16 + h].
// ---------------------------------------------------------------------------
__global__ void prep_w_kernel(const float* __restrict__ W, _Float16* __restrict__ wfrag) {
    int i = blockIdx.x * blockDim.x + threadIdx.x;     // 0 .. 24575
    if (i >= KSTEPS * NTILES * 32 * 16) return;
    int h  = i & 15;
    int l  = (i >> 4) & 31;
    int nt = (i >> 9) & 3;
    int s  = i >> 11;
    int n  = l & 15;
    int hi = l >> 4;
    int k  = s * 32 + ((h & 8) << 1) + hi * 8 + (h & 7);
    wfrag[i] = (_Float16)W[(nt * 16 + n) * FEAT + k];
}

// ---------------------------------------------------------------------------
// Main: one wave per 32-node tile (two 16-row M-tiles share each B fragment).
// ---------------------------------------------------------------------------
__global__ __launch_bounds__(WAVES_PER_BLOCK * 32)
void graphconv_kernel(const float* __restrict__ x,          // [GNODES, 64]
                      const int* __restrict__ nbr,          // [GNODES, 5]
                      const _Float16* __restrict__ wfrag,   // [48*512] f16
                      const float* __restrict__ bias,       // [64]
                      float* __restrict__ out) {            // [GNODES, 64]
    __shared__ _Float16 feat[WAVES_PER_BLOCK][TILE_M][ROW_STRIDE];  // 50176 B

    const int lane  = threadIdx.x & 31;
    const int wave  = threadIdx.x >> 5;
    const int tile  = blockIdx.x * WAVES_PER_BLOCK + wave;  // 0 .. 12499
    const int gbase = tile * TILE_M;
    const int half  = lane >> 4;        // half-wave id (also A-frag K-chunk select)
    const int qlane = lane & 15;

    _Float16 (*frow)[ROW_STRIDE] = feat[wave];

    // ---- Phase 1: gather. Each half-wave loads one full 64-float row per
    // instruction (16 lanes x float4), converts to f16, stores 8B to LDS.
    // Half-wave h covers rows h*16 .. h*16+15.
    for (int rq = 0; rq < 16; ++rq) {
        const int r = half * 16 + rq;
        const int g = gbase + r;
        #pragma unroll
        for (int seg = 0; seg < 6; ++seg) {
            int src = (seg == 0) ? g : nbr[g * GK + (seg - 1)];
            v4f f = *(const v4f*)(x + (size_t)src * GF + qlane * 4);
            v4h hv;
            hv[0] = (_Float16)f[0];
            hv[1] = (_Float16)f[1];
            hv[2] = (_Float16)f[2];
            hv[3] = (_Float16)f[3];
            *(v4h*)&frow[r][seg * GF + qlane * 4] = hv;
        }
    }
    __syncthreads();

    // ---- Phase 2: 12 K-steps x (2 M-tiles x 4 N-tiles) = 96 WMMAs/wave.
    // Each B fragment is loaded once and reused by both M-tiles.
    v8f acc[MTILES][NTILES];
    #pragma unroll
    for (int mt = 0; mt < MTILES; ++mt)
        #pragma unroll
        for (int nt = 0; nt < NTILES; ++nt) acc[mt][nt] = (v8f){};

    const _Float16* arow0 = frow[qlane];         // M-tile 0: rows 0..15
    const _Float16* arow1 = frow[16 + qlane];    // M-tile 1: rows 16..31
    const _Float16* wbase = wfrag + (size_t)lane * 16;

    for (int s = 0; s < KSTEPS; ++s) {
        // ISA 16-bit A 16x32 layout: lane holds row; lanes<16 K-chunks {0..7,16..23},
        // lanes>=16 K-chunks {8..15,24..31} of the 32-wide step.
        v16h a[MTILES];
        #pragma unroll
        for (int mt = 0; mt < MTILES; ++mt) {
            const _Float16* ar = (mt == 0) ? arow0 : arow1;
            v8h alo = *(const v8h*)(ar + s * 32 + half * 8);
            v8h ahi = *(const v8h*)(ar + s * 32 + 16 + half * 8);
            #pragma unroll
            for (int i = 0; i < 8; ++i) { a[mt][i] = alo[i]; a[mt][8 + i] = ahi[i]; }
        }
        #pragma unroll
        for (int nt = 0; nt < NTILES; ++nt) {
            v16h bf = *(const v16h*)(wbase + (size_t)(s * NTILES + nt) * 512);
            acc[0][nt] = __builtin_amdgcn_wmma_f32_16x16x32_f16(
                false, a[0], false, bf, (short)0, acc[0][nt], false, false);
            acc[1][nt] = __builtin_amdgcn_wmma_f32_16x16x32_f16(
                false, a[1], false, bf, (short)0, acc[1][nt], false, false);
        }
    }

    // ---- Epilogue: bias + fp32 store.
    // C layout: lane<16 -> M=r, lane>=16 -> M=8+r; N = lane&15.
    #pragma unroll
    for (int nt = 0; nt < NTILES; ++nt) {
        const int o   = nt * 16 + qlane;
        const float bv = bias[o];
        #pragma unroll
        for (int mt = 0; mt < MTILES; ++mt) {
            #pragma unroll
            for (int r = 0; r < 8; ++r) {
                const int mm = mt * 16 + r + 8 * half;
                out[(size_t)(gbase + mm) * GOUT + o] = acc[mt][nt][r] + bv;
            }
        }
    }
}

// ---------------------------------------------------------------------------
extern "C" void kernel_launch(void* const* d_in, const int* in_sizes, int n_in,
                              void* d_out, int out_size, void* d_ws, size_t ws_size,
                              hipStream_t stream) {
    const float* x    = (const float*)d_in[0];   // [B, N, F] fp32
    const int*   nbr  = (const int*)d_in[1];     // [B, N, K] int
    const float* W    = (const float*)d_in[2];   // [OUT, 6F] fp32
    const float* bias = (const float*)d_in[3];   // [OUT] fp32
    float* out        = (float*)d_out;           // [B, N, OUT] fp32
    _Float16* wfrag   = (_Float16*)d_ws;         // 48 fragments * 512 halves = 48KB

    (void)in_sizes; (void)n_in; (void)out_size; (void)ws_size;

    // Pre-swizzle W into WMMA B-fragment layout (tiny, runs every call).
    const int welems = KSTEPS * NTILES * 32 * 16;  // 24576
    prep_w_kernel<<<(welems + 255) / 256, 256, 0, stream>>>(W, wfrag);

    // 400000 nodes / 32 per wave = 12500 waves; 2 waves/block -> 6250 blocks.
    const int tiles  = GNODES / TILE_M;
    const int blocks = tiles / WAVES_PER_BLOCK;
    graphconv_kernel<<<blocks, WAVES_PER_BLOCK * 32, 0, stream>>>(x, nbr, wfrag, bias, out);
}
